// MainModelAttention_11003706212635
// MI455X (gfx1250) — compile-verified, matrix operand order checked by reading, not run on hardware
//
#include <hip/hip_runtime.h>
#include <hip/hip_bf16.h>

typedef __bf16 bf16_t;
typedef __attribute__((ext_vector_type(16))) __bf16       v16bf;
typedef __attribute__((ext_vector_type(8)))  float        v8f;
typedef __attribute__((ext_vector_type(4)))  unsigned int u32x4;
typedef __attribute__((ext_vector_type(4)))  int          v4i;

#define BM 256          // block rows
#define BN 128          // block cols
#define BK 32           // K step
#define LDS_STRIDE 40   // 32 + 8 bf16 pad (80B rows)

// CDNA5 async global->LDS path (ASYNCcnt), guarded so the build can't regress.
#if defined(__has_builtin)
#if __has_builtin(__builtin_amdgcn_global_load_async_to_lds_b128) && \
    __has_builtin(__builtin_amdgcn_s_wait_asynccnt)
#define USE_ASYNC_LDS 1
#endif
#endif
#ifndef USE_ASYNC_LDS
#define USE_ASYNC_LDS 0
#endif

#if USE_ASYNC_LDS
typedef __attribute__((address_space(1))) v4i* glb_v4i_p;  // pointer to AS1 int4
typedef __attribute__((address_space(3))) v4i* lds_v4i_p;  // pointer to AS3 int4
#endif

__device__ __forceinline__ void cp_b128_to_lds(bf16_t* dst_lds, const bf16_t* src_glb) {
#if USE_ASYNC_LDS
    __builtin_amdgcn_global_load_async_to_lds_b128(
        (glb_v4i_p)(src_glb), (lds_v4i_p)(dst_lds), 0, 0);
#else
    *(u32x4*)dst_lds = *(const u32x4*)src_glb;
#endif
}

__device__ __forceinline__ void wait_stage_done() {
#if USE_ASYNC_LDS
    __builtin_amdgcn_s_wait_asynccnt(0);
#endif
}

// ---------------------------------------------------------------------------
// f32 -> bf16 convert (grid-stride)
// ---------------------------------------------------------------------------
__global__ __launch_bounds__(256) void cvt_f32_to_bf16(const float* __restrict__ in,
                                                       bf16_t* __restrict__ out, long n) {
    long i = (long)blockIdx.x * blockDim.x + threadIdx.x;
    long stride = (long)gridDim.x * blockDim.x;
    for (; i < n; i += stride) out[i] = (bf16_t)in[i];
}

// ---------------------------------------------------------------------------
// Generic bf16 WMMA GEMM:  C[z] = A[z] (MxK row-major) * Bt[z]^T (Bt NxK
// row-major, second operand K-major) + bias
// OUT_MODE: 0 = f32 row-major, 1 = bf16 row-major, 2 = bf16 transposed
// 256 threads = 8 waves (4x2). Block tile 256x128, wave tile 64x64, BK=32.
// LDS double-buffered; one barrier per K step; async global->LDS staging.
// ---------------------------------------------------------------------------
template <int OUT_MODE>
__global__ __launch_bounds__(256) void gemm_bf16_wmma(
    const bf16_t* __restrict__ A,  long aBatch, int lda,
    const bf16_t* __restrict__ Bt, long bBatch, int ldb,
    const float*  __restrict__ bias,
    void* __restrict__ out, long oBatch, int ldo,
    int K)
{
    __shared__ bf16_t As[2][BM][LDS_STRIDE];
    __shared__ bf16_t Bs[2][BN][LDS_STRIDE];

    const int tid  = threadIdx.x;
    const int lane = tid & 31;
    const int wave = tid >> 5;
    const int wm   = wave & 3;      // 4 wave-rows of 64
    const int wn   = wave >> 2;     // 2 wave-cols of 64
    const int half = lane >> 4;     // K-half select per WMMA 16-bit layout
    const int l15  = lane & 15;

    const int blockM = blockIdx.y * BM;
    const int blockN = blockIdx.x * BN;
    const int z      = blockIdx.z;

    const bf16_t* Ab = A  + (long)z * aBatch;
    const bf16_t* Bb = Bt + (long)z * bBatch;

    v8f acc[4][4];
    const v8f zero8 = {0.f, 0.f, 0.f, 0.f, 0.f, 0.f, 0.f, 0.f};
#pragma unroll
    for (int i = 0; i < 4; ++i)
#pragma unroll
        for (int j = 0; j < 4; ++j) acc[i][j] = zero8;

    union Frag { v16bf v; u32x4 q[2]; };

    // Prologue: stage tile 0 into buffer 0.
    // A tile: 256x32 bf16 = 1024 u32x4 chunks (4/thread); B tile: 512 (2/thread)
#pragma unroll
    for (int it = 0; it < 4; ++it) {
        int chunk = tid + it * 256;
        int row = chunk >> 2, c16 = (chunk & 3) * 8;
        cp_b128_to_lds(&As[0][row][c16], Ab + (long)(blockM + row) * lda + c16);
    }
#pragma unroll
    for (int it = 0; it < 2; ++it) {
        int chunk = tid + it * 256;
        int row = chunk >> 2, c16 = (chunk & 3) * 8;
        cp_b128_to_lds(&Bs[0][row][c16], Bb + (long)(blockN + row) * ldb + c16);
    }
    wait_stage_done();
    __syncthreads();

    for (int kb = 0; kb < K; kb += BK) {
        const int cur = (kb >> 5) & 1;
        const int nxt = cur ^ 1;

        // Stage next tile into the other buffer while computing on this one.
        if (kb + BK < K) {
#pragma unroll
            for (int it = 0; it < 4; ++it) {
                int chunk = tid + it * 256;
                int row = chunk >> 2, c16 = (chunk & 3) * 8;
                cp_b128_to_lds(&As[nxt][row][c16],
                               Ab + (long)(blockM + row) * lda + (kb + BK) + c16);
            }
#pragma unroll
            for (int it = 0; it < 2; ++it) {
                int chunk = tid + it * 256;
                int row = chunk >> 2, c16 = (chunk & 3) * 8;
                cp_b128_to_lds(&Bs[nxt][row][c16],
                               Bb + (long)(blockN + row) * ldb + (kb + BK) + c16);
            }
        }

        // Fragments per CDNA5 16-bit WMMA layout:
        // lane<16 holds K 0..7 / 16..23 ; lane>=16 holds K 8..15 / 24..31
        Frag a[4], b[4];
#pragma unroll
        for (int i = 0; i < 4; ++i) {
            int r = wm * 64 + i * 16 + l15;
            a[i].q[0] = *(const u32x4*)&As[cur][r][half * 8];
            a[i].q[1] = *(const u32x4*)&As[cur][r][half * 8 + 16];
        }
#pragma unroll
        for (int j = 0; j < 4; ++j) {
            int c = wn * 64 + j * 16 + l15;
            b[j].q[0] = *(const u32x4*)&Bs[cur][c][half * 8];
            b[j].q[1] = *(const u32x4*)&Bs[cur][c][half * 8 + 16];
        }

#pragma unroll
        for (int i = 0; i < 4; ++i)
#pragma unroll
            for (int j = 0; j < 4; ++j)
                acc[i][j] = __builtin_amdgcn_wmma_f32_16x16x32_bf16(
                    false, a[i].v, false, b[j].v, (short)0, acc[i][j], false, false);

        wait_stage_done();     // this wave's async stage of `nxt` complete
        __syncthreads();       // all waves: `nxt` filled, `cur` reads retired
    }

    // Epilogue. C/D layout: VGPR r -> row = base + half*8 + r, col = base + l15
#pragma unroll
    for (int i = 0; i < 4; ++i) {
#pragma unroll
        for (int j = 0; j < 4; ++j) {
            int col = blockN + wn * 64 + j * 16 + l15;
            float bv = bias ? bias[col] : 0.f;
#pragma unroll
            for (int r = 0; r < 8; ++r) {
                int row = blockM + wm * 64 + i * 16 + half * 8 + r;
                float v = acc[i][j][r] + bv;
                if (OUT_MODE == 0) {
                    ((float*)out)[(long)z * oBatch + (long)row * ldo + col] = v;
                } else if (OUT_MODE == 1) {
                    ((bf16_t*)out)[(long)z * oBatch + (long)row * ldo + col] = (bf16_t)v;
                } else {
                    ((bf16_t*)out)[(long)z * oBatch + (long)col * ldo + row] = (bf16_t)v;
                }
            }
        }
    }
}

// ---------------------------------------------------------------------------
// Row softmax with scale + additive mask; f32 scores -> bf16 probs
// ---------------------------------------------------------------------------
__global__ __launch_bounds__(256) void softmax_rows(
    const float* __restrict__ scores, const float* __restrict__ mask,
    bf16_t* __restrict__ probs, int S, float scale)
{
    const int row = blockIdx.x;
    const int b   = blockIdx.y;
    const float* srow = scores + ((long)b * S + row) * (long)S;
    const float* mrow = mask + (long)b * S;
    bf16_t* prow = probs + ((long)b * S + row) * (long)S;

    __shared__ float red[256];
    const int tid = threadIdx.x;

    float m = -3.0e38f;
    for (int c = tid; c < S; c += 256)
        m = fmaxf(m, srow[c] * scale + mrow[c]);
    red[tid] = m;
    __syncthreads();
    for (int off = 128; off > 0; off >>= 1) {
        if (tid < off) red[tid] = fmaxf(red[tid], red[tid + off]);
        __syncthreads();
    }
    m = red[0];
    __syncthreads();

    float s = 0.f;
    for (int c = tid; c < S; c += 256)
        s += __expf(srow[c] * scale + mrow[c] - m);
    red[tid] = s;
    __syncthreads();
    for (int off = 128; off > 0; off >>= 1) {
        if (tid < off) red[tid] += red[tid + off];
        __syncthreads();
    }
    const float inv = 1.f / red[0];

    for (int c = tid; c < S; c += 256)
        prow[c] = (bf16_t)(__expf(srow[c] * scale + mrow[c] - m) * inv);
}

// ---------------------------------------------------------------------------
extern "C" void kernel_launch(void* const* d_in, const int* in_sizes, int n_in,
                              void* d_out, int out_size, void* d_ws, size_t ws_size,
                              hipStream_t stream) {
    const float* hidden = (const float*)d_in[0];
    const float* mask   = (const float*)d_in[1];
    const float* Wq     = (const float*)d_in[2];
    const float* bq     = (const float*)d_in[3];
    const float* Wk     = (const float*)d_in[4];
    const float* bk     = (const float*)d_in[5];
    const float* Wv     = (const float*)d_in[6];
    const float* bv     = (const float*)d_in[7];
    float* out = (float*)d_out;

    const int B = 2, S = 4096, H = 2048;
    const long BS   = (long)B * S;   // 8192
    const long NH   = (long)H * H;   // 4.2M
    const long NBSH = BS * H;        // 16.7M

    char* ws = (char*)d_ws;
    size_t off = 0;
    bf16_t* Hbf  = (bf16_t*)(ws + off); off += (size_t)NBSH * 2;
    bf16_t* Qbf  = (bf16_t*)(ws + off); off += (size_t)NBSH * 2;
    bf16_t* Kbf  = (bf16_t*)(ws + off); off += (size_t)NBSH * 2;
    bf16_t* Vt   = (bf16_t*)(ws + off); off += (size_t)NBSH * 2;  // [H, B*S]
    bf16_t* Wqbf = (bf16_t*)(ws + off); off += (size_t)NH * 2;
    bf16_t* Wkbf = (bf16_t*)(ws + off); off += (size_t)NH * 2;
    bf16_t* Wvbf = (bf16_t*)(ws + off); off += (size_t)NH * 2;
    float*  Sc   = (float*) (ws + off); off += (size_t)B * S * (size_t)S * 4;
    bf16_t* Pbf  = (bf16_t*)(ws + off); off += (size_t)B * S * (size_t)S * 2;

    dim3 blk(256);

    // 1) convert inputs to bf16
    cvt_f32_to_bf16<<<2048, blk, 0, stream>>>(hidden, Hbf, NBSH);
    cvt_f32_to_bf16<<<1024, blk, 0, stream>>>(Wq, Wqbf, NH);
    cvt_f32_to_bf16<<<1024, blk, 0, stream>>>(Wk, Wkbf, NH);
    cvt_f32_to_bf16<<<1024, blk, 0, stream>>>(Wv, Wvbf, NH);

    // 2) projections: [B*S,H] @ [H,H]^T, batches folded into M
    dim3 gProj(H / BN, BS / BM, 1);
    gemm_bf16_wmma<1><<<gProj, blk, 0, stream>>>(Hbf, 0, H, Wqbf, 0, H, bq,
                                                 Qbf, 0, H, H);
    gemm_bf16_wmma<1><<<gProj, blk, 0, stream>>>(Hbf, 0, H, Wkbf, 0, H, bk,
                                                 Kbf, 0, H, H);
    // V written transposed: Vt[h][b*S + s]  (ldo = B*S)
    gemm_bf16_wmma<2><<<gProj, blk, 0, stream>>>(Hbf, 0, H, Wvbf, 0, H, bv,
                                                 Vt, 0, (int)BS, H);

    // 3) scores[b] = Q[b] @ K[b]^T  (raw f32; scale+mask folded into softmax)
    dim3 gScore(S / BN, S / BM, B);
    gemm_bf16_wmma<0><<<gScore, blk, 0, stream>>>(Qbf, (long)S * H, H,
                                                  Kbf, (long)S * H, H, nullptr,
                                                  Sc, (long)S * S, S, H);

    // 4) softmax rows -> bf16 probs
    dim3 gSm(S, B);
    softmax_rows<<<gSm, blk, 0, stream>>>(Sc, mask, Pbf, S,
                                          0.08838834764831845f /* 1/sqrt(128) */);

    // 5) context[b] = P[b] @ V[b]  (V K-major via Vt: per-batch offset b*S in
    //    each [H, B*S] row, row stride B*S)
    dim3 gCtx(H / BN, S / BM, B);
    gemm_bf16_wmma<0><<<gCtx, blk, 0, stream>>>(Pbf, (long)S * S, S,
                                                Vt, (long)S, (int)BS, nullptr,
                                                out, (long)S * H, H, S);
}
